// BaselineGCN_51470888075302
// MI455X (gfx1250) — matrix operand order repaired and checked by reading, not compiled, not run on hardware
//
#include <hip/hip_runtime.h>

typedef __attribute__((ext_vector_type(16))) __bf16 v16bf;
typedef __attribute__((ext_vector_type(8)))  float  v8f;

#define FEAT 128

__device__ __forceinline__ void splitbf(float x, __bf16& hi, __bf16& lo) {
    __bf16 h = (__bf16)x;
    hi = h;
    lo = (__bf16)(x - (float)h);
}

// ---------------------------------------------------------------------------
// Fused dense layer front-end:
//   tmp = act(h) @ W          (act = ReLU for layers >= 1, identity for layer 0)
//   out = bias + tmp * dis^2  (self-loop term, so scatter only handles edges)
// fp32-accurate via bf16 hi/lo split (Ah*Bh + Al*Bh + Ah*Bl).
// One wave -> 16x128 strip; W staged in LDS pre-swizzled into B-frag layout.
// ---------------------------------------------------------------------------
__global__ __launch_bounds__(256) void gcn_gemm_fused(
    const float* __restrict__ h, const float* __restrict__ W,
    const float* __restrict__ bias, const float* __restrict__ dis,
    float* __restrict__ tmp, float* __restrict__ out,
    int nrows, int apply_relu)
{
    __shared__ alignas(64) __bf16 Whi[FEAT * FEAT];
    __shared__ alignas(64) __bf16 Wlo[FEAT * FEAT];

    // Cooperative load of W, converted to bf16 hi/lo and swizzled so that a
    // B fragment (v16bf per lane) is one contiguous 32B LDS read.
    for (int i = threadIdx.x; i < FEAT * FEAT; i += 256) {
        int k = i >> 7, n = i & 127;
        float w = W[i];
        __bf16 hi, lo; splitbf(w, hi, lo);
        int ki   = k >> 5;                        // K-step (32 per step)
        int c    = n >> 4;                        // N-tile
        int lane = (((k >> 3) & 1) << 4) | (n & 15);
        int j    = (k & 7) | (((k >> 4) & 1) << 3); // element within fragment
        int idx  = ((ki * 8 + c) << 9) + (lane << 4) + j;
        Whi[idx] = hi; Wlo[idx] = lo;
    }
    __syncthreads();

    int wave  = threadIdx.x >> 5;
    int lane  = threadIdx.x & 31;
    int strip = blockIdx.x * 8 + wave;
    int row0  = strip * 16;
    if (row0 >= nrows) return;                    // wave-uniform

    int  lrow = row0 + (lane & 15);
    bool rok  = lrow < nrows;
    const float* hrow = h + (size_t)lrow * FEAT;
    int kb = (lane >> 4) << 3;                    // 0 or 8 (A-frag K base)

    v8f zero = {0.f,0.f,0.f,0.f,0.f,0.f,0.f,0.f};
    v8f acc[8];
    #pragma unroll
    for (int c = 0; c < 8; ++c) acc[c] = zero;

    #pragma unroll
    for (int ki = 0; ki < 4; ++ki) {
        int k0 = ki * 32;
        // Build A fragment (16x32 bf16) per ISA layout, hi/lo split.
        v16bf ahi, alo;
        #pragma unroll
        for (int j = 0; j < 8; ++j) {
            float x = rok ? hrow[k0 + kb + j] : 0.0f;          // K = kb..kb+7
            if (apply_relu) x = fmaxf(x, 0.0f);
            __bf16 hi, lo; splitbf(x, hi, lo);
            ahi[j] = hi; alo[j] = lo;
            float y = rok ? hrow[k0 + kb + 16 + j] : 0.0f;     // K = kb+16..kb+23
            if (apply_relu) y = fmaxf(y, 0.0f);
            splitbf(y, hi, lo);
            ahi[8 + j] = hi; alo[8 + j] = lo;
        }
        #pragma unroll
        for (int c = 0; c < 8; ++c) {
            const v16bf bh = *(const v16bf*)&Whi[((ki * 8 + c) << 9) + (lane << 4)];
            const v16bf bl = *(const v16bf*)&Wlo[((ki * 8 + c) << 9) + (lane << 4)];
            acc[c] = __builtin_amdgcn_wmma_f32_16x16x32_bf16(
                false, ahi, false, bh, (short)0, acc[c], false, false);
            acc[c] = __builtin_amdgcn_wmma_f32_16x16x32_bf16(
                false, alo, false, bh, (short)0, acc[c], false, false);
            acc[c] = __builtin_amdgcn_wmma_f32_16x16x32_bf16(
                false, ahi, false, bl, (short)0, acc[c], false, false);
        }
    }

    // Epilogue. C layout: VGPR r -> M = r + 8*(lane>=16), N = lane&15 (+16c).
    int rbase = row0 + ((lane >> 4) << 3);
    int cbase = lane & 15;

    float dsq[8];
    #pragma unroll
    for (int r = 0; r < 8; ++r) {
        int row = rbase + r;
        float d = (row < nrows) ? dis[row] : 0.0f;
        dsq[r] = d * d;
    }
    #pragma unroll
    for (int c = 0; c < 8; ++c) {
        int   col = c * 16 + cbase;
        float b   = bias[col];
        #pragma unroll
        for (int r = 0; r < 8; ++r) {
            int row = rbase + r;
            if (row < nrows) {
                float v = acc[c][r];
                size_t idx = (size_t)row * FEAT + col;
                tmp[idx] = v;                     // messages for edge scatter
                out[idx] = b + v * dsq[r];        // bias + self-loop term
            }
        }
    }
}

// ---------------------------------------------------------------------------
// Degree / normalization
// ---------------------------------------------------------------------------
__global__ void gcn_deg_init(float* deg, int n) {
    int i = blockIdx.x * 256 + threadIdx.x;
    if (i < n) deg[i] = 1.0f;                     // self loop
}
__global__ void gcn_deg_count(const long long* __restrict__ dst, float* deg, int E) {
    int e = blockIdx.x * 256 + threadIdx.x;
    if (e < E) unsafeAtomicAdd(&deg[(int)dst[e]], 1.0f);
}
__global__ void gcn_dis(float* deg, int n) {
    int i = blockIdx.x * 256 + threadIdx.x;
    if (i < n) {
        float d = deg[i];
        deg[i] = d > 0.0f ? rsqrtf(d) : 0.0f;
    }
}

// ---------------------------------------------------------------------------
// Edge scatter: one wave per edge; lane -> float4 feature chunk; fp32 atomics.
// Edge id forced wave-uniform so src/dst/norm become scalar loads.
// ---------------------------------------------------------------------------
__global__ void gcn_scatter(const float* __restrict__ tmp, const float* __restrict__ dis,
                            const long long* __restrict__ src, const long long* __restrict__ dst,
                            float* __restrict__ out, int E) {
    int gid = blockIdx.x * 256 + threadIdx.x;
    int e = __builtin_amdgcn_readfirstlane(gid >> 5);   // wave-uniform edge id
    if (e >= E) return;
    int lane = gid & 31;
    int s = (int)src[e];
    int d = (int)dst[e];
    float w = dis[s] * dis[d];
    float4 v = ((const float4*)(tmp + (size_t)s * FEAT))[lane];
    float* ob = out + (size_t)d * FEAT + lane * 4;
    unsafeAtomicAdd(ob + 0, v.x * w);
    unsafeAtomicAdd(ob + 1, v.y * w);
    unsafeAtomicAdd(ob + 2, v.z * w);
    unsafeAtomicAdd(ob + 3, v.w * w);
}

// ---------------------------------------------------------------------------
// Final projection with fused ReLU: out[n] = relu(h[n,:]) . lin_w + lin_b
// ---------------------------------------------------------------------------
__global__ void gcn_final(const float* __restrict__ h, const float* __restrict__ w,
                          const float* __restrict__ b, float* __restrict__ out, int n) {
    int i = blockIdx.x * 256 + threadIdx.x;
    if (i >= n) return;
    const float4* hv = (const float4*)(h + (size_t)i * FEAT);
    const float4* wv = (const float4*)w;
    float acc = 0.0f;
    #pragma unroll 8
    for (int j = 0; j < FEAT / 4; ++j) {
        float4 a = hv[j], c = wv[j];
        acc += fmaxf(a.x, 0.f) * c.x + fmaxf(a.y, 0.f) * c.y
             + fmaxf(a.z, 0.f) * c.z + fmaxf(a.w, 0.f) * c.w;
    }
    out[i] = acc + b[0];
}

// ---------------------------------------------------------------------------
extern "C" void kernel_launch(void* const* d_in, const int* in_sizes, int n_in,
                              void* d_out, int out_size, void* d_ws, size_t ws_size,
                              hipStream_t stream) {
    const float*     x   = (const float*)d_in[0];
    const long long* ei  = (const long long*)d_in[1];   // int64 edge_index
    const float*     Ws  = (const float*)d_in[2];
    const float*     bs  = (const float*)d_in[3];
    const float*     lw  = (const float*)d_in[4];
    const float*     lb  = (const float*)d_in[5];

    int N = in_sizes[0] / FEAT;
    int E = in_sizes[1] / 2;
    const long long* src = ei;
    const long long* dst = ei + E;

    // Workspace: dis[N] | bufA[N*128] (gemm output tmp) | bufB[N*128] (h ping-pong)
    float* dis = (float*)d_ws;
    size_t off = (((size_t)N * sizeof(float)) + 255) & ~(size_t)255;
    float* bufA = (float*)((char*)d_ws + off);
    float* bufB = bufA + (size_t)N * FEAT;

    int nb = (N + 255) / 256;
    gcn_deg_init <<<nb, 256, 0, stream>>>(dis, N);
    gcn_deg_count<<<(E + 255) / 256, 256, 0, stream>>>(dst, dis, E);
    gcn_dis      <<<nb, 256, 0, stream>>>(dis, N);

    int strips  = (N + 15) / 16;
    int gblocks = (strips + 7) / 8;
    int sb      = (E * 32 + 255) / 256;

    const float* h = x;
    for (int l = 0; l < 7; ++l) {
        gcn_gemm_fused<<<gblocks, 256, 0, stream>>>(
            h, Ws + (size_t)l * FEAT * FEAT, bs + l * FEAT, dis,
            bufA, bufB, N, l > 0 ? 1 : 0);
        gcn_scatter<<<sb, 256, 0, stream>>>(bufA, dis, src, dst, bufB, E);
        h = bufB;
    }
    gcn_final<<<nb, 256, 0, stream>>>(bufB, lw, lb, (float*)d_out, N);
}
